// CausalSelfAttention_60017872995077
// MI455X (gfx1250) — matrix operand order repaired and checked by reading, not compiled, not run on hardware
//
#include <hip/hip_runtime.h>
#include <math.h>

// ---------------------------------------------------------------------------
// CDNA5 (gfx1250) wave32 WMMA implementation of causal self-attention block.
// All GEMM-shaped work runs on v_wmma_f32_16x16x32_f16 (fp32 accum).
// B=2, T=2048, C=2048, H=16, D=128 per the reference.
// ---------------------------------------------------------------------------

typedef __attribute__((ext_vector_type(16))) _Float16 v16h;
typedef __attribute__((ext_vector_type(8)))  _Float16 v8h;
typedef __attribute__((ext_vector_type(8)))  float    v8f;

#define NHEAD 16
#define SEQ_T 2048   // reference config (T not derivable from flat sizes)

// --- WMMA fragment helpers (layouts per CDNA5 ISA 7.12.2, wave32) -----------

// A fragment, 16x32 f16, tile origin p=(row0,k0), row-major with leading dim ld.
// lane L (m=L&15, hl=L>>4): halves [k = hl*8 + 0..7] then [k = 16 + hl*8 + 0..7]
static __device__ __forceinline__ v16h load_frag_a(const _Float16* p, int ld) {
    const int lane = threadIdx.x & 31;
    const int m = lane & 15, hl = lane >> 4;
    const _Float16* r0 = p + (size_t)m * ld + hl * 8;
    union { v16h v; v8h h[2]; } u;
    u.h[0] = *(const v8h*)(r0);
    u.h[1] = *(const v8h*)(r0 + 16);
    return u.v;
}

// B fragment, 32x16 f16, sourced from a row-major [N,K] array (i.e. op^T),
// tile origin p=(n0,k0). lane L (n=L&15, hl=L>>4): halves [k = hl*8 + 0..7] then [k = 16 + hl*8 + 0..7]
static __device__ __forceinline__ v16h load_frag_b(const _Float16* p, int ld) {
    const int lane = threadIdx.x & 31;
    const int n = lane & 15, hl = lane >> 4;
    const _Float16* r0 = p + (size_t)n * ld + hl * 8;
    union { v16h v; v8h h[2]; } u;
    u.h[0] = *(const v8h*)(r0);
    u.h[1] = *(const v8h*)(r0 + 16);
    return u.v;
}

static __device__ __forceinline__ v8f wmma_f16(v16h a, v16h b, v8f c) {
    return __builtin_amdgcn_wmma_f32_16x16x32_f16(false, a, false, b, (short)0, c,
                                                  false, false);
}

// --- elementwise kernels ----------------------------------------------------

__global__ void cast_f32_to_f16(const float* __restrict__ src,
                                _Float16* __restrict__ dst, size_t n) {
    for (size_t i = (size_t)blockIdx.x * blockDim.x + threadIdx.x; i < n;
         i += (size_t)gridDim.x * blockDim.x)
        dst[i] = (_Float16)src[i];
}

// depthwise causal conv (k=3) + cast to f16.  out[b,t,c] = w0*x[t-2]+w1*x[t-1]+w2*x[t]
__global__ void conv_cast_kernel(const float* __restrict__ x,
                                 const float* __restrict__ w,
                                 _Float16* __restrict__ xo,
                                 int B, int T, int C) {
    size_t idx = (size_t)blockIdx.x * blockDim.x + threadIdx.x;
    size_t total = (size_t)B * T * C;
    if (idx >= total) return;
    int c = (int)(idx % C);
    size_t bt = idx / C;
    int t = (int)(bt % T);
    size_t b = bt / T;
    const float* xb = x + b * (size_t)T * C;
    float acc = w[c * 3 + 2] * xb[(size_t)t * C + c];
    if (t >= 1) acc += w[c * 3 + 1] * xb[(size_t)(t - 1) * C + c];
    if (t >= 2) acc += w[c * 3 + 0] * xb[(size_t)(t - 2) * C + c];
    xo[idx] = (_Float16)acc;
}

// RoPE on q,k (+ fold 1/sqrt(D) into q) and scatter qkv -> [B,H,T,D] (q,k)
// and v -> transposed [B,H,D,T] so the P@V WMMA B-operand is key-contiguous.
__global__ void rope_scatter_kernel(const _Float16* __restrict__ qkv,
                                    _Float16* __restrict__ Qo,
                                    _Float16* __restrict__ Ko,
                                    _Float16* __restrict__ Vt,
                                    int B, int T, int C, int H, int D,
                                    float qscale) {
    size_t idx = (size_t)blockIdx.x * blockDim.x + threadIdx.x;
    size_t total = (size_t)B * T * C;
    if (idx >= total) return;
    int c = (int)(idx % C);
    size_t bt = idx / C;
    int t = (int)(bt % T);
    int b = (int)(bt / T);
    int h = c / D, d = c % D;
    const _Float16* base = qkv + bt * (size_t)(3 * C);

    float qv = (float)base[c];
    float kv = (float)base[C + c];
    float vv = (float)base[2 * C + c];

    int i = d & (D / 2 - 1);                       // d % (D/2)
    float inv = __expf(-(logf(10000.0f) * (2.0f * (float)i) / (float)D));
    float ang = (float)t * inv;
    float sn, cs;
    __sincosf(ang, &sn, &cs);
    int dp = (d < D / 2) ? d + D / 2 : d - D / 2;  // rotate_half partner
    float sgn = (d < D / 2) ? -1.0f : 1.0f;
    float qp = (float)base[h * D + dp];
    float kp = (float)base[C + h * D + dp];

    float qo = (qv * cs + sgn * qp * sn) * qscale;
    float ko = kv * cs + sgn * kp * sn;

    size_t bhtd = (((size_t)b * H + h) * T + t) * (size_t)D + d;
    Qo[bhtd] = (_Float16)qo;
    Ko[bhtd] = (_Float16)ko;
    Vt[(((size_t)b * H + h) * D + d) * (size_t)T + t] = (_Float16)vv;
}

// --- WMMA GEMM:  out[M,N] = A[M,K] * Bt[N,K]^T  ("NT", both K-contiguous) ---
// block = 256 threads = 8 waves, wave tile 32x64 (2x4 fragments), block 128x128
__global__ __launch_bounds__(256) void gemm_nt_kernel(
    const _Float16* __restrict__ A, const _Float16* __restrict__ Bt,
    _Float16* __restrict__ outH, float* __restrict__ outF,
    int M, int N, int K) {
    const int wid  = threadIdx.x >> 5;
    const int lane = threadIdx.x & 31;
    const int wm = wid >> 1, wn = wid & 1;
    const int row0 = blockIdx.y * 128 + wm * 32;
    const int col0 = blockIdx.x * 128 + wn * 64;

    v8f acc[2][4] = {};
    for (int k0 = 0; k0 < K; k0 += 32) {
        v16h a0 = load_frag_a(A + (size_t)row0 * K + k0, K);
        v16h a1 = load_frag_a(A + (size_t)(row0 + 16) * K + k0, K);
#pragma unroll
        for (int j = 0; j < 4; ++j) {
            v16h b = load_frag_b(Bt + (size_t)(col0 + j * 16) * K + k0, K);
            acc[0][j] = wmma_f16(a0, b, acc[0][j]);
            acc[1][j] = wmma_f16(a1, b, acc[1][j]);
        }
    }
    const int n = lane & 15, hl = lane >> 4;
#pragma unroll
    for (int i = 0; i < 2; ++i)
#pragma unroll
        for (int j = 0; j < 4; ++j)
#pragma unroll
            for (int r = 0; r < 8; ++r) {
                int rr = row0 + i * 16 + r + 8 * hl;
                int cc = col0 + j * 16 + n;
                float v = acc[i][j][r];
                if (outH) outH[(size_t)rr * N + cc] = (_Float16)v;
                else      outF[(size_t)rr * N + cc] = v;
            }
}

// --- flash attention: block = 4 waves, wave owns 16 q-rows, 32-key tiles ----
__global__ __launch_bounds__(128) void attn_kernel(
    const _Float16* __restrict__ Q,   // [B*H, T, D]  (pre-scaled by 1/sqrt(D))
    const _Float16* __restrict__ Km,  // [B*H, T, D]
    const _Float16* __restrict__ Vt,  // [B*H, D, T]
    _Float16* __restrict__ Y,         // [B, T, C] f16 (feeds proj GEMM)
    int T, int H, int D, int C) {
    __shared__ _Float16 plds[4][16][32];   // per-wave P re-layout scratch

    const int wid  = threadIdx.x >> 5;
    const int lane = threadIdx.x & 31;
    const int nloc = lane & 15, hl = lane >> 4;
    const int bh = blockIdx.y;
    const int b = bh / H, h = bh % H;
    const int qr0 = blockIdx.x * 64 + wid * 16;

    const _Float16* q  = Q  + (size_t)bh * T * D;
    const _Float16* kp = Km + (size_t)bh * T * D;
    const _Float16* vt = Vt + (size_t)bh * D * T;

    v16h aq[4];
#pragma unroll
    for (int kk = 0; kk < 4; ++kk)
        aq[kk] = load_frag_a(q + (size_t)qr0 * D + kk * 32, D);

    v8f o[8] = {};
    float mrow[8], lrow[8];
#pragma unroll
    for (int r = 0; r < 8; ++r) { mrow[r] = -3.0e38f; lrow[r] = 0.0f; }

    const int ktEnd = (qr0 + 15) >> 5;
    for (int kt = 0; kt <= ktEnd; ++kt) {
        const int key0 = kt * 32;
        // S = q @ k^T for 16 rows x 32 keys (two 16x16 fragments, K=128 -> 4 steps)
        v8f s0 = {}, s1 = {};
#pragma unroll
        for (int kk = 0; kk < 4; ++kk) {
            v16h b0 = load_frag_b(kp + (size_t)key0 * D + kk * 32, D);
            v16h b1 = load_frag_b(kp + (size_t)(key0 + 16) * D + kk * 32, D);
            s0 = wmma_f16(aq[kk], b0, s0);
            s1 = wmma_f16(aq[kk], b1, s1);
        }
        // online softmax with causal mask; rows live per-VGPR, reduce across
        // the 16-lane half-wave holding each row.
        const int kc0 = key0 + nloc, kc1 = kc0 + 16;
#pragma unroll
        for (int r = 0; r < 8; ++r) {
            const int row = qr0 + r + 8 * hl;
            float x0 = (kc0 <= row) ? s0[r] : -3.0e38f;
            float x1 = (kc1 <= row) ? s1[r] : -3.0e38f;
            float mx = fmaxf(x0, x1);
#pragma unroll
            for (int off = 1; off < 16; off <<= 1)
                mx = fmaxf(mx, __shfl_xor(mx, off, 32));
            float mnew = fmaxf(mrow[r], mx);
            float corr = __expf(mrow[r] - mnew);
            float p0 = __expf(x0 - mnew);
            float p1 = __expf(x1 - mnew);
            float ps = p0 + p1;
#pragma unroll
            for (int off = 1; off < 16; off <<= 1)
                ps += __shfl_xor(ps, off, 32);
            lrow[r] = lrow[r] * corr + ps;
            mrow[r] = mnew;
#pragma unroll
            for (int nf = 0; nf < 8; ++nf) o[nf][r] *= corr;
            plds[wid][r + 8 * hl][nloc]      = (_Float16)p0;
            plds[wid][r + 8 * hl][nloc + 16] = (_Float16)p1;
        }
        // same-wave LDS RAW: explicit CDNA5 split-counter wait
        asm volatile("s_wait_dscnt 0" ::: "memory");
        // re-gather P in A-fragment layout, then O += P @ V (8 d-tiles)
        union { v16h v; v8h hh[2]; } u;
        const _Float16* pr = &plds[wid][nloc][hl * 8];
        u.hh[0] = *(const v8h*)pr;
        u.hh[1] = *(const v8h*)(pr + 16);
        v16h ap = u.v;
#pragma unroll
        for (int nf = 0; nf < 8; ++nf) {
            v16h bv = load_frag_b(vt + (size_t)(nf * 16) * T + key0, T);
            o[nf] = wmma_f16(ap, bv, o[nf]);
        }
    }
    // epilogue: normalize rows, write f16 y[B,T,C] for the proj GEMM
#pragma unroll
    for (int r = 0; r < 8; ++r) {
        const int row = qr0 + r + 8 * hl;
        const float inv = 1.0f / lrow[r];
#pragma unroll
        for (int nf = 0; nf < 8; ++nf) {
            float vy = o[nf][r] * inv;
            Y[((size_t)b * T + row) * C + h * D + nf * 16 + nloc] = (_Float16)vy;
        }
    }
}

// ---------------------------------------------------------------------------

extern "C" void kernel_launch(void* const* d_in, const int* in_sizes, int n_in,
                              void* d_out, int out_size, void* d_ws, size_t ws_size,
                              hipStream_t stream) {
    const float* x      = (const float*)d_in[0];
    const float* conv_w = (const float*)d_in[1];
    const float* w_attn = (const float*)d_in[2];
    const float* w_proj = (const float*)d_in[3];
    float* out = (float*)d_out;

    const int C = in_sizes[1] / 3;          // conv_w is (C,1,3)
    const int T = SEQ_T;
    const int B = (int)((size_t)in_sizes[0] / ((size_t)T * C));
    const int H = NHEAD, D = C / H;
    const int M = B * T;

    // workspace carve-up (all f16)
    char* ws = (char*)d_ws;
    _Float16* wAttnH = (_Float16*)ws; ws += (size_t)3 * C * C * sizeof(_Float16);
    _Float16* wProjH = (_Float16*)ws; ws += (size_t)C * C * sizeof(_Float16);
    _Float16* xconvH = (_Float16*)ws; ws += (size_t)M * C * sizeof(_Float16);
    _Float16* qkvH   = (_Float16*)ws; ws += (size_t)M * 3 * C * sizeof(_Float16);
    _Float16* qH     = (_Float16*)ws; ws += (size_t)M * C * sizeof(_Float16);
    _Float16* kH     = (_Float16*)ws; ws += (size_t)M * C * sizeof(_Float16);
    _Float16* vT     = (_Float16*)ws; ws += (size_t)M * C * sizeof(_Float16);
    _Float16* yH     = (_Float16*)ws; ws += (size_t)M * C * sizeof(_Float16);

    // 1) weight casts to f16
    cast_f32_to_f16<<<2048, 256, 0, stream>>>(w_attn, wAttnH, (size_t)3 * C * C);
    cast_f32_to_f16<<<2048, 256, 0, stream>>>(w_proj, wProjH, (size_t)C * C);

    // 2) depthwise causal conv + cast
    {
        size_t total = (size_t)M * C;
        int blocks = (int)((total + 255) / 256);
        conv_cast_kernel<<<blocks, 256, 0, stream>>>(x, conv_w, xconvH, B, T, C);
    }

    // 3) QKV GEMM: [M,C] x [3C,C]^T -> [M,3C]  (f16 WMMA, fp32 accum)
    gemm_nt_kernel<<<dim3((3 * C) / 128, M / 128), 256, 0, stream>>>(
        xconvH, wAttnH, qkvH, nullptr, M, 3 * C, C);

    // 4) RoPE (+1/sqrt(D) into q) and scatter to attention layouts
    {
        size_t total = (size_t)M * C;
        int blocks = (int)((total + 255) / 256);
        float qscale = 1.0f / sqrtf((float)D);
        rope_scatter_kernel<<<blocks, 256, 0, stream>>>(qkvH, qH, kH, vT,
                                                        B, T, C, H, D, qscale);
    }

    // 5) causal flash attention (WMMA for QK^T and PV)
    attn_kernel<<<dim3(T / 64, B * H), 128, 0, stream>>>(qH, kH, vT, yH,
                                                         T, H, D, C);

    // 6) output projection: [M,C] x [C,C]^T -> fp32 d_out
    gemm_nt_kernel<<<dim3(C / 128, M / 128), 256, 0, stream>>>(
        yH, wProjH, nullptr, out, M, C, C);
}
